// InfiniAttention_3178275799212
// MI455X (gfx1250) — matrix operand order repaired and checked by reading, not compiled, not run on hardware
//
#include <hip/hip_runtime.h>

// ---------------------------------------------------------------------------
// InfiniAttention for MI455X (gfx1250, wave32, WMMA bf16 path)
// B=2, S=4096, HID=2048, H=32, D=64, NB=8, BS=512, MEM=64, THETA=10000
// ---------------------------------------------------------------------------

typedef __attribute__((ext_vector_type(16))) __bf16 v16bf;
typedef __attribute__((ext_vector_type(16))) unsigned short v16us;
typedef __attribute__((ext_vector_type(8))) float v8f;

union Frag16 {
  v16us u16;
  unsigned int u32[8];
};

__device__ __forceinline__ unsigned short f2bf(float f) {
  unsigned int u = __float_as_uint(f);
  u += 0x7FFFu + ((u >> 16) & 1u);  // round-to-nearest-even
  return (unsigned short)(u >> 16);
}
__device__ __forceinline__ float bf2f(unsigned short h) {
  return __uint_as_float(((unsigned int)h) << 16);
}

__device__ __forceinline__ v8f wmma_bf16(const Frag16& a, const Frag16& b, v8f c) {
  // D = A(16x32 bf16) * B(32x16 bf16) + C(16x16 f32)
  return __builtin_amdgcn_wmma_f32_16x16x32_bf16(
      false, __builtin_bit_cast(v16bf, a.u16),
      false, __builtin_bit_cast(v16bf, b.u16),
      (short)0, c, false, false);
}

// --------------------------- async copy support ----------------------------
#if __has_builtin(__builtin_amdgcn_global_load_async_to_lds_b128)
#define HAVE_ASYNC_LDS 1
#else
#define HAVE_ASYNC_LDS 0
#endif

// Builtin expects GCC-style v4i pointers: AS(1) source, AS(3) destination.
typedef int v4i_gcc __attribute__((vector_size(16)));
typedef __attribute__((address_space(1))) v4i_gcc* gvec_t;
typedef __attribute__((address_space(3))) v4i_gcc* lvec_t;

__device__ __forceinline__ gvec_t as_g(const void* p) {
  return (gvec_t)(unsigned long long)(size_t)p;  // generic and AS1 share 64-bit repr
}
__device__ __forceinline__ lvec_t as_l(void* p) {
  // ISA: flat LDS address low 32 bits == LDS byte offset
  return (lvec_t)(unsigned int)(size_t)p;
}
__device__ __forceinline__ void wait_async0() {
#if __has_builtin(__builtin_amdgcn_s_wait_asynccnt)
  __builtin_amdgcn_s_wait_asynccnt(0);
#else
  asm volatile("s_wait_asynccnt 0" ::: "memory");
#endif
}

// ---------------------------------------------------------------------------
// Kernel 1: fp32 -> bf16 convert (grid-stride)
// ---------------------------------------------------------------------------
__global__ __launch_bounds__(256) void cvt_f32_bf16(const float* __restrict__ in,
                                                    unsigned short* __restrict__ out,
                                                    int n) {
  int i = blockIdx.x * blockDim.x + threadIdx.x;
  int stride = gridDim.x * blockDim.x;
  for (; i < n; i += stride) out[i] = f2bf(in[i]);
}

// ---------------------------------------------------------------------------
// Kernel 1b: memory_k -> bf16 [H,MEM,D]; memory_v -> bf16 transposed [H,D,MEM]
// ---------------------------------------------------------------------------
__global__ __launch_bounds__(256) void cvt_mem(const float* __restrict__ mk,
                                               const float* __restrict__ mv,
                                               unsigned short* __restrict__ mkb,
                                               unsigned short* __restrict__ mvt) {
  int i = blockIdx.x * blockDim.x + threadIdx.x;  // 32*64*64 threads
  int h = i >> 12, m = (i >> 6) & 63, d = i & 63;
  mkb[i] = f2bf(mk[i]);
  mvt[(h << 12) + (d << 6) + m] = f2bf(mv[i]);
}

// ---------------------------------------------------------------------------
// Kernel 2/5: GEMM  C[M,N] = A[M,K] * B[N,K]^T   (both row-major bf16)
// Tiles: 128x128x64, 256 threads (8 waves), 16 WMMAs/wave per K-stage.
// Double-buffered LDS staging via GLOBAL_LOAD_ASYNC_TO_LDS_B128 (ASYNCcnt).
// EPI=0: store f32 to C.  EPI=1: scatter bf16 into q/k [B,H,S,D], v [B,H,D,S].
// ---------------------------------------------------------------------------
constexpr int BM = 128, BN = 128, BK = 64;
constexpr int SA = 72, SB = 72;  // LDS row stride (u16), 16B-aligned rows + pad

template <int EPI>
__global__ __launch_bounds__(256) void gemm_bf16_nt(
    const unsigned short* __restrict__ A, const unsigned short* __restrict__ B,
    float* __restrict__ C, unsigned short* __restrict__ qb,
    unsigned short* __restrict__ kb, unsigned short* __restrict__ vb,
    int M, int N, int K) {
  __shared__ unsigned short As[2][BM * SA];
  __shared__ unsigned short Bs[2][BN * SB];
  const int tid = threadIdx.x;
  const int lane = tid & 31;
  const int wid = tid >> 5;
  const int r = lane & 15;
  const int hb = lane >> 4;        // lane half (wave32)
  const int wm = (wid >> 1) * 32;  // wave row offset within 128
  const int wn = (wid & 1) * 64;   // wave col offset within 128
  const int srow = tid >> 1;       // staging: 2 threads/row, 32 u16 each
  const int scol = (tid & 1) * 32;
  const size_t abase = (size_t)blockIdx.x * BM;
  const size_t bbase = (size_t)blockIdx.y * BN;

  auto issue = [&](int buf, int k0) {
    const unsigned short* ga = A + (abase + srow) * K + k0 + scol;
    const unsigned short* gb = B + (bbase + srow) * K + k0 + scol;
    unsigned short* la = &As[buf][srow * SA + scol];
    unsigned short* lb = &Bs[buf][srow * SB + scol];
#if HAVE_ASYNC_LDS
#pragma unroll
    for (int c = 0; c < 4; ++c) {
      __builtin_amdgcn_global_load_async_to_lds_b128(as_g(ga + 8 * c), as_l(la + 8 * c), 0, 0);
      __builtin_amdgcn_global_load_async_to_lds_b128(as_g(gb + 8 * c), as_l(lb + 8 * c), 0, 0);
    }
#else
    const uint4* ag = reinterpret_cast<const uint4*>(ga);
    const uint4* bg = reinterpret_cast<const uint4*>(gb);
#pragma unroll
    for (int c = 0; c < 4; ++c) {
      uint4 av = ag[c];
      uint4 bv = bg[c];
      *reinterpret_cast<uint4*>(la + 8 * c) = av;
      *reinterpret_cast<uint4*>(lb + 8 * c) = bv;
    }
#endif
  };

  v8f acc[2][4];
#pragma unroll
  for (int i = 0; i < 2; ++i)
#pragma unroll
    for (int j = 0; j < 4; ++j)
#pragma unroll
      for (int e = 0; e < 8; ++e) acc[i][j][e] = 0.0f;

  const int nk = K / BK;
  issue(0, 0);
  for (int it = 0; it < nk; ++it) {
#if HAVE_ASYNC_LDS
    wait_async0();
#endif
    __syncthreads();
    if (it + 1 < nk) issue((it + 1) & 1, (it + 1) * BK);

    const unsigned short* Asb = As[it & 1];
    const unsigned short* Bsb = Bs[it & 1];
#pragma unroll
    for (int ks = 0; ks < 2; ++ks) {  // two 32-wide K sub-steps per stage
      const int ko = ks * 32;
      Frag16 af[2], bf[4];
#pragma unroll
      for (int t = 0; t < 2; ++t) {
        const unsigned short* as = &Asb[(wm + t * 16 + r) * SA + ko];
#pragma unroll
        for (int j = 0; j < 8; ++j) {
          int kk = hb * 16 + 2 * j;  // A 16x32 layout
          af[t].u32[j] = *reinterpret_cast<const unsigned int*>(&as[kk]);
        }
      }
#pragma unroll
      for (int t = 0; t < 4; ++t) {
        const unsigned short* bs = &Bsb[(wn + t * 16 + r) * SB + ko];
#pragma unroll
        for (int j = 0; j < 8; ++j) {
          int kk = hb * 16 + 2 * j;  // B 32x16 layout
          bf[t].u32[j] = *reinterpret_cast<const unsigned int*>(&bs[kk]);
        }
      }
#pragma unroll
      for (int i = 0; i < 2; ++i)
#pragma unroll
        for (int j = 0; j < 4; ++j) acc[i][j] = wmma_bf16(af[i], bf[j], acc[i][j]);
    }
    __syncthreads();
  }

#pragma unroll
  for (int i = 0; i < 2; ++i)
#pragma unroll
    for (int j = 0; j < 4; ++j)
#pragma unroll
      for (int e = 0; e < 8; ++e) {
        int gm = (int)abase + wm + i * 16 + hb * 8 + e;  // C layout rows
        int gn = (int)bbase + wn + j * 16 + r;           // C layout cols
        float v = acc[i][j][e];
        if (EPI == 0) {
          C[(size_t)gm * N + gn] = v;
        } else {
          int bb = gm >> 12, s = gm & 4095;  // M = B*S rows
          int which = gn >> 11, rr = gn & 2047;
          int h = rr >> 6, d = rr & 63;
          unsigned short bv = f2bf(v);
          if (which == 0) {
            qb[(((size_t)bb * 32 + h) * 4096 + s) * 64 + d] = bv;
          } else if (which == 1) {
            kb[(((size_t)bb * 32 + h) * 4096 + s) * 64 + d] = bv;
          } else {  // V stored transposed: [B,H,D,S]
            vb[(((size_t)bb * 32 + h) * 64 + d) * 4096 + s] = bv;
          }
        }
      }
}

// ---------------------------------------------------------------------------
// Kernel 3: RoPE in place on q,k (bf16, [B,H,S,D]); pairs (d, d+32).
// Q additionally folded with the 1/sqrt(D)=0.125 score scale (exact in bf16).
// ---------------------------------------------------------------------------
__global__ __launch_bounds__(256) void rope_kernel(unsigned short* __restrict__ q,
                                                   unsigned short* __restrict__ k) {
  unsigned int i = blockIdx.x * blockDim.x + threadIdx.x;  // B*H*S*32 threads
  int pair = i & 31;
  int s = (i >> 5) & 4095;
  int bh = i >> 17;
  size_t base = ((size_t)bh * 4096 + s) * 64;
  float invf = __powf(10000.0f, -(float)(2 * pair) * (1.0f / 64.0f));
  float ang = (float)s * invf;
  float c = __cosf(ang), sn = __sinf(ang);
  {
    float x1 = bf2f(q[base + pair]);
    float x2 = bf2f(q[base + pair + 32]);
    q[base + pair] = f2bf((x1 * c - x2 * sn) * 0.125f);
    q[base + pair + 32] = f2bf((x2 * c + x1 * sn) * 0.125f);
  }
  {
    float x1 = bf2f(k[base + pair]);
    float x2 = bf2f(k[base + pair + 32]);
    k[base + pair] = f2bf(x1 * c - x2 * sn);
    k[base + pair + 32] = f2bf(x2 * c + x1 * sn);
  }
}

// ---------------------------------------------------------------------------
// Kernel 4: fused blockwise-causal flash attention + memory attention + gate.
// One block (8 waves) per (b, h, 128-query tile); each wave owns 16 q rows.
// V is [B,H,D,S] so P*V B-fragments load contiguously; rowsums via WMMA.
// ---------------------------------------------------------------------------
__global__ __launch_bounds__(256) void attn_kernel(
    const unsigned short* __restrict__ qg, const unsigned short* __restrict__ kg,
    const unsigned short* __restrict__ vt, const float* __restrict__ gate,
    const unsigned short* __restrict__ mkb, const unsigned short* __restrict__ mvt,
    unsigned short* __restrict__ ctx) {
  __shared__ unsigned short Ps[8 * 16 * 72];  // per-wave 16x64 (stride 72) pad
  const int tid = threadIdx.x, lane = tid & 31, wid = tid >> 5;
  const int r = lane & 15, hb = lane >> 4;
  const int blk = blockIdx.x;
  const int qt = blk & 31;
  const int h = (blk >> 5) & 31;
  const int b = blk >> 10;
  const int qr0 = qt * 128 + wid * 16;  // absolute first query row (s)
  const size_t bho = ((size_t)b * 32 + h) * 4096 * 64;
  const unsigned short* Q = qg + bho;
  const unsigned short* Kp = kg + bho;
  const unsigned short* VT = vt + bho;  // [D][S] within (b,h)
  unsigned short* myP = &Ps[wid * 16 * 72];

  Frag16 ones;
#pragma unroll
  for (int j = 0; j < 8; ++j) ones.u32[j] = 0x3F803F80u;  // bf16 1.0 pair

  // Q fragments (16x64 -> two 16x32 A frags); already scaled by 0.125
  Frag16 qa[2];
#pragma unroll
  for (int cc = 0; cc < 2; ++cc) {
    const unsigned short* qs = Q + (size_t)(qr0 + r) * 64 + cc * 32;
#pragma unroll
    for (int j = 0; j < 8; ++j) {
      int kk = hb * 16 + 2 * j;
      qa[cc].u32[j] = *reinterpret_cast<const unsigned int*>(qs + kk);
    }
  }

  float m_i[8], l_i[8];
  v8f o[4];
#pragma unroll
  for (int j = 0; j < 8; ++j) { m_i[j] = -1e30f; l_i[j] = 0.0f; }
#pragma unroll
  for (int t = 0; t < 4; ++t)
#pragma unroll
    for (int e = 0; e < 8; ++e) o[t][e] = 0.0f;

  const int bstart = (qr0 >> 9) << 9;  // 512-block start
  for (int kc = bstart; kc < qr0 + 16; kc += 32) {
    if (kc + 32 < qr0 + 16) {  // prefetch next chunk (global_prefetch_b8)
      __builtin_prefetch(Kp + (size_t)(kc + 32 + r) * 64, 0, 0);
      __builtin_prefetch(VT + (size_t)(r << 2) * 4096 + kc + 32, 0, 0);
    }
    // S = Q * K^T for a 32-key chunk (two 16x16 C frags)
    v8f sc[2];
#pragma unroll
    for (int kt = 0; kt < 2; ++kt) {
      v8f sa;
#pragma unroll
      for (int e = 0; e < 8; ++e) sa[e] = 0.0f;
#pragma unroll
      for (int cc = 0; cc < 2; ++cc) {
        Frag16 bk;
        const unsigned short* ks = Kp + (size_t)(kc + kt * 16 + r) * 64 + cc * 32;
#pragma unroll
        for (int j = 0; j < 8; ++j) {
          int kk = hb * 16 + 2 * j;
          bk.u32[j] = *reinterpret_cast<const unsigned int*>(ks + kk);
        }
        sa = wmma_bf16(qa[cc], bk, sa);
      }
      sc[kt] = sa;
    }
    // V fragments hoisted here: contiguous from vT, overlap with softmax VALU
    Frag16 bvv[4];
#pragma unroll
    for (int t = 0; t < 4; ++t) {
      const unsigned short* vs = VT + (size_t)(t * 16 + r) * 4096 + kc + hb * 16;
#pragma unroll
      for (int j = 0; j < 8; ++j)
        bvv[t].u32[j] = *reinterpret_cast<const unsigned int*>(vs + 2 * j);
    }
    // causal mask + chunk row-max (Q pre-scaled, no mul here)
    float cmax[8];
#pragma unroll
    for (int j = 0; j < 8; ++j) cmax[j] = -1e30f;
#pragma unroll
    for (int kt = 0; kt < 2; ++kt)
#pragma unroll
      for (int j = 0; j < 8; ++j) {
        int key = kc + kt * 16 + r;
        int qrow = qr0 + hb * 8 + j;
        float v = sc[kt][j];
        if (key > qrow) v = -1e30f;
        sc[kt][j] = v;
        cmax[j] = fmaxf(cmax[j], v);
      }
#pragma unroll
    for (int j = 0; j < 8; ++j)
#pragma unroll
      for (int msk = 1; msk < 16; msk <<= 1)
        cmax[j] = fmaxf(cmax[j], __shfl_xor(cmax[j], msk, 32));
    // online softmax update (row sums via WMMA below)
    float alpha[8];
#pragma unroll
    for (int j = 0; j < 8; ++j) {
      float mn = fmaxf(m_i[j], cmax[j]);
      alpha[j] = __expf(m_i[j] - mn);
      m_i[j] = mn;
    }
#pragma unroll
    for (int kt = 0; kt < 2; ++kt)
#pragma unroll
      for (int j = 0; j < 8; ++j) sc[kt][j] = __expf(sc[kt][j] - m_i[j]);
#pragma unroll
    for (int t = 0; t < 4; ++t)
#pragma unroll
      for (int j = 0; j < 8; ++j) o[t][j] *= alpha[j];
    // P (C-layout) -> LDS -> A-layout bf16 fragment
#pragma unroll
    for (int kt = 0; kt < 2; ++kt)
#pragma unroll
      for (int j = 0; j < 8; ++j)
        myP[(hb * 8 + j) * 72 + kt * 16 + r] = f2bf(sc[kt][j]);
    asm volatile("s_wait_dscnt 0" ::: "memory");
    Frag16 pa;
#pragma unroll
    for (int j = 0; j < 8; ++j) {
      int kk = hb * 16 + 2 * j;
      pa.u32[j] = *reinterpret_cast<const unsigned int*>(&myP[r * 72 + kk]);
    }
    // rowsum(P) = P x ones : lands in C layout, per-lane VGPR j == row j sum
    v8f rsf;
#pragma unroll
    for (int e = 0; e < 8; ++e) rsf[e] = 0.0f;
    rsf = wmma_bf16(pa, ones, rsf);
#pragma unroll
    for (int j = 0; j < 8; ++j) l_i[j] = l_i[j] * alpha[j] + rsf[j];
    // O += P * V
#pragma unroll
    for (int t = 0; t < 4; ++t) o[t] = wmma_bf16(pa, bvv[t], o[t]);
  }

  // ---- memory attention over MEM=64 slots (single-pass softmax) ----
  const unsigned short* MK = mkb + (h << 12);   // [MEM][D] bf16
  const unsigned short* MVT = mvt + (h << 12);  // [D][MEM] bf16
  v8f sm[4];
#pragma unroll
  for (int mt = 0; mt < 4; ++mt) {
    v8f sa;
#pragma unroll
    for (int e = 0; e < 8; ++e) sa[e] = 0.0f;
#pragma unroll
    for (int cc = 0; cc < 2; ++cc) {
      Frag16 bm;
      const unsigned short* ms = MK + (size_t)(mt * 16 + r) * 64 + cc * 32;
#pragma unroll
      for (int j = 0; j < 8; ++j) {
        int kk = hb * 16 + 2 * j;
        bm.u32[j] = *reinterpret_cast<const unsigned int*>(ms + kk);
      }
      sa = wmma_bf16(qa[cc], bm, sa);
    }
    sm[mt] = sa;
  }
  float mm[8];
#pragma unroll
  for (int j = 0; j < 8; ++j) mm[j] = -1e30f;
#pragma unroll
  for (int mt = 0; mt < 4; ++mt)
#pragma unroll
    for (int j = 0; j < 8; ++j) mm[j] = fmaxf(mm[j], sm[mt][j]);
#pragma unroll
  for (int j = 0; j < 8; ++j)
#pragma unroll
    for (int msk = 1; msk < 16; msk <<= 1)
      mm[j] = fmaxf(mm[j], __shfl_xor(mm[j], msk, 32));
#pragma unroll
  for (int mt = 0; mt < 4; ++mt)
#pragma unroll
    for (int j = 0; j < 8; ++j) sm[mt][j] = __expf(sm[mt][j] - mm[j]);
  // P_mem (16x64) via LDS transpose
#pragma unroll
  for (int mt = 0; mt < 4; ++mt)
#pragma unroll
    for (int j = 0; j < 8; ++j)
      myP[(hb * 8 + j) * 72 + mt * 16 + r] = f2bf(sm[mt][j]);
  asm volatile("s_wait_dscnt 0" ::: "memory");
  Frag16 pm[2];
#pragma unroll
  for (int cc = 0; cc < 2; ++cc)
#pragma unroll
    for (int j = 0; j < 8; ++j) {
      int kk = cc * 32 + hb * 16 + 2 * j;
      pm[cc].u32[j] = *reinterpret_cast<const unsigned int*>(&myP[r * 72 + kk]);
    }
  // denominator via WMMA rowsum
  v8f lsf;
#pragma unroll
  for (int e = 0; e < 8; ++e) lsf[e] = 0.0f;
  lsf = wmma_bf16(pm[0], ones, lsf);
  lsf = wmma_bf16(pm[1], ones, lsf);
  v8f om[4];
#pragma unroll
  for (int t = 0; t < 4; ++t)
#pragma unroll
    for (int e = 0; e < 8; ++e) om[t][e] = 0.0f;
#pragma unroll
  for (int t = 0; t < 4; ++t) {
#pragma unroll
    for (int cc = 0; cc < 2; ++cc) {
      Frag16 bv;
      const unsigned short* mvs = MVT + (size_t)(t * 16 + r) * 64 + cc * 32 + hb * 16;
#pragma unroll
      for (int j = 0; j < 8; ++j)
        bv.u32[j] = *reinterpret_cast<const unsigned int*>(mvs + 2 * j);
      om[t] = wmma_bf16(pm[cc], bv, om[t]);
    }
  }
  // ---- gate + store ctx [B,S,HID] bf16 ----
  float g = 1.0f / (1.0f + __expf(-gate[h]));
#pragma unroll
  for (int t = 0; t < 4; ++t)
#pragma unroll
    for (int j = 0; j < 8; ++j) {
      int row = hb * 8 + j;
      int ss = qr0 + row;
      int d = t * 16 + r;
      float lc = o[t][j] / l_i[j];
      float mc = om[t][j] / lsf[j];
      ctx[((size_t)b * 4096 + ss) * 2048 + h * 64 + d] = f2bf((1.0f - g) * lc + g * mc);
    }
}

// ---------------------------------------------------------------------------
// Host launcher
// ---------------------------------------------------------------------------
extern "C" void kernel_launch(void* const* d_in, const int* in_sizes, int n_in,
                              void* d_out, int out_size, void* d_ws, size_t ws_size,
                              hipStream_t stream) {
  (void)in_sizes; (void)n_in; (void)out_size; (void)ws_size;
  const float* x = (const float*)d_in[0];
  const float* w_qkv = (const float*)d_in[1];
  const float* w_o = (const float*)d_in[2];
  const float* gate = (const float*)d_in[3];
  const float* mem_k = (const float*)d_in[4];
  const float* mem_v = (const float*)d_in[5];
  float* out = (float*)d_out;

  const size_t nX = (size_t)8192 * 2048;           // B*S*HID
  const size_t nWQ = (size_t)6144 * 2048;          // 3H*D x HID
  const size_t nWO = (size_t)2048 * 2048;
  const size_t nQKV = (size_t)2 * 32 * 4096 * 64;  // per q/k/v
  const size_t nMEM = (size_t)32 * 64 * 64;        // H*MEM*D

  unsigned short* ws = (unsigned short*)d_ws;
  unsigned short* x_bf = ws;
  unsigned short* wqkv_bf = x_bf + nX;
  unsigned short* wo_bf = wqkv_bf + nWQ;
  unsigned short* q_bf = wo_bf + nWO;
  unsigned short* k_bf = q_bf + nQKV;
  unsigned short* vT_bf = k_bf + nQKV;  // [B,H,D,S]
  unsigned short* ctx_bf = vT_bf + nQKV;
  unsigned short* mk_bf = ctx_bf + nX;    // [H,MEM,D]
  unsigned short* mvT_bf = mk_bf + nMEM;  // [H,D,MEM]

  cvt_f32_bf16<<<2048, 256, 0, stream>>>(x, x_bf, (int)nX);
  cvt_f32_bf16<<<2048, 256, 0, stream>>>(w_qkv, wqkv_bf, (int)nWQ);
  cvt_f32_bf16<<<1024, 256, 0, stream>>>(w_o, wo_bf, (int)nWO);
  cvt_mem<<<512, 256, 0, stream>>>(mem_k, mem_v, mk_bf, mvT_bf);

  dim3 g1(64, 48);  // M=8192/128, N=6144/128
  gemm_bf16_nt<1><<<g1, 256, 0, stream>>>(x_bf, wqkv_bf, nullptr, q_bf, k_bf, vT_bf,
                                          8192, 6144, 2048);

  rope_kernel<<<32768, 256, 0, stream>>>(q_bf, k_bf);  // 2*32*4096*32 threads

  attn_kernel<<<2048, 256, 0, stream>>>(q_bf, k_bf, vT_bf, gate, mk_bf, mvT_bf, ctx_bf);

  dim3 g2(64, 16);  // M=8192/128, N=2048/128
  gemm_bf16_nt<0><<<g2, 256, 0, stream>>>(ctx_bf, wo_bf, out, nullptr, nullptr, nullptr,
                                          8192, 2048, 2048);
}
